// UI_Aggregator_79998060855420
// MI455X (gfx1250) — compile-verified, matrix operand order checked by reading, not run on hardware
//
#include <hip/hip_runtime.h>

// ---------------------------------------------------------------------------
// UI_Aggregator for MI455X (gfx1250, wave32, WMMA)
//   Live math only (entmax over a singleton axis == 1 -> attention MLP dead,
//   both heads identical):
//     S[b]   = sum_l l2norm( a*e_ui + (1-a)*e_r ),  a = sigmoid(X @ gate_w + b)
//     out    = sigmoid-gated mix of BN/MLP(S) with u2e_w[nodes]
// ---------------------------------------------------------------------------

#define B_N   4096
#define L_N   50
#define D_N   64
#define TD    192          // 3*D
#define EPSN  1e-12f
#define BN_EPS 1e-5f

#define XS 200             // hX row stride (halves), 16B-aligned, conflict-free
#define WTS 200            // hWt row stride (halves)

typedef __attribute__((ext_vector_type(16))) _Float16 v16h;
typedef __attribute__((ext_vector_type(8)))  _Float16 v8h;
typedef __attribute__((ext_vector_type(8)))  float    v8f;

// ---------------------------------------------------------------------------
// Kernel 1: gather + gate GEMM (WMMA f16->f32) + gated fusion + l2norm + sum_l
// One workgroup per b; 128 threads = 4 waves; rows 0..63 (50 real, 14 padded).
// ---------------------------------------------------------------------------
__global__ __launch_bounds__(128)
void k_gate_agg(const int* __restrict__ hui, const int* __restrict__ hr,
                const float* __restrict__ i2e, const float* __restrict__ r2e,
                const float* __restrict__ gate_w, const float* __restrict__ gate_b,
                float* __restrict__ S)
{
    __shared__ __align__(16) _Float16 hWt[64 * WTS];  // gate_w^T, f16
    __shared__ __align__(16) _Float16 hX [64 * XS];   // [e_ui | e_r | e_ui*e_r] f16; last slot reused for o
    __shared__ float sGB[64];
    __shared__ float red[128];
    __shared__ float rnorm[64];

    const int b = blockIdx.x;
    const int t = threadIdx.x;

    // --- stage gate_w (k-major [192][64]) transposed into hWt[n][k] as f16 ---
    for (int idx = t; idx < TD * D_N; idx += 128) {
        int k = idx >> 6, n = idx & 63;
        hWt[n * WTS + k] = (_Float16)gate_w[idx];
    }
    if (t < 64) sGB[t] = gate_b[t];

    // --- gather embeddings, build X rows in f16 (zero-pad rows 50..63) ---
    {
        int m = t >> 1;            // 0..63
        int half = t & 1;
        int c0 = half * 32;
        if (m < L_N) {
            int ri = hui[b * L_N + m];
            int rr = hr [b * L_N + m];
            const float4* pi = (const float4*)(i2e + (size_t)ri * D_N + c0);
            const float4* pr = (const float4*)(r2e + (size_t)rr * D_N + c0);
            #pragma unroll
            for (int q = 0; q < 8; ++q) {
                float4 a = pi[q];
                float4 c = pr[q];
                float va[4] = {a.x, a.y, a.z, a.w};
                float vc[4] = {c.x, c.y, c.z, c.w};
                #pragma unroll
                for (int s2 = 0; s2 < 4; ++s2) {
                    int j = c0 + q * 4 + s2;
                    float eu = va[s2], er = vc[s2];
                    hX[m * XS + j]       = (_Float16)eu;
                    hX[m * XS + 64 + j]  = (_Float16)er;
                    hX[m * XS + 128 + j] = (_Float16)(eu * er);
                }
            }
        } else {
            for (int j = c0; j < c0 + 32; ++j) {
                hX[m * XS + j]       = (_Float16)0.f;
                hX[m * XS + 64 + j]  = (_Float16)0.f;
                hX[m * XS + 128 + j] = (_Float16)0.f;
            }
        }
    }
    __syncthreads();

    // --- WMMA: [16,192] x [192,64] per wave; 4 N-tiles x 6 K-chunks ---
    {
        const int wave  = t >> 5;
        const int lane  = t & 31;
        const int mf    = lane & 15;          // M (A) / N (B) within tile
        const int kb    = (lane >> 4) * 8;    // K sub-offset per lane half
        const int mrow0 = wave * 16;

        v8f acc[4];
        #pragma unroll
        for (int nt = 0; nt < 4; ++nt) acc[nt] = (v8f){};

        #pragma unroll
        for (int nt = 0; nt < 4; ++nt) {
            #pragma unroll
            for (int c = 0; c < 6; ++c) {
                const _Float16* ap = &hX[(mrow0 + mf) * XS + c * 32 + kb];
                v8h alo = *(const v8h*)ap;
                v8h ahi = *(const v8h*)(ap + 16);
                v16h af = __builtin_shufflevector(alo, ahi,
                          0,1,2,3,4,5,6,7,8,9,10,11,12,13,14,15);
                const _Float16* bp = &hWt[(nt * 16 + mf) * WTS + c * 32 + kb];
                v8h blo = *(const v8h*)bp;
                v8h bhi = *(const v8h*)(bp + 16);
                v16h bf = __builtin_shufflevector(blo, bhi,
                          0,1,2,3,4,5,6,7,8,9,10,11,12,13,14,15);
                acc[nt] = __builtin_amdgcn_wmma_f32_16x16x32_f16(
                              false, af, false, bf, (short)0, acc[nt], false, false);
            }
        }

        // --- fuse: a = sigmoid(acc + bias); o = a*e_ui + (1-a)*e_r ---
        // Write o (f16) into the e_ui*e_r slot (done with it; own rows only).
        const int mbase = mrow0 + (lane >> 4) * 8;
        #pragma unroll
        for (int nt = 0; nt < 4; ++nt) {
            int j = nt * 16 + mf;
            float bias = sGB[j];
            #pragma unroll
            for (int v = 0; v < 8; ++v) {
                int m = mbase + v;
                float pre = acc[nt][v] + bias;
                float a = 1.f / (1.f + __expf(-pre));
                float eu = (float)hX[m * XS + j];
                float er = (float)hX[m * XS + 64 + j];
                hX[m * XS + 128 + j] = (_Float16)(a * eu + (1.f - a) * er);
            }
        }
    }
    __syncthreads();

    // --- per-row L2 norms of o ---
    {
        int m = t >> 1, half = t & 1;
        float ss = 0.f;
        for (int j = half * 32; j < half * 32 + 32; ++j) {
            float o = (float)hX[m * XS + 128 + j];
            ss += o * o;
        }
        red[t] = ss;
    }
    __syncthreads();
    if (t < 64) {
        float nrm = sqrtf(red[2 * t] + red[2 * t + 1]);
        rnorm[t] = 1.f / fmaxf(nrm, EPSN);    // zero rows -> o==0, contribute 0
    }
    __syncthreads();

    // --- column sums of normalized o -> S[b] ---
    {
        int j = t & 63, rs = t >> 6;
        float s = 0.f;
        for (int m = rs * 32; m < rs * 32 + 32; ++m)
            s += (float)hX[m * XS + 128 + j] * rnorm[m];
        red[t] = s;
    }
    __syncthreads();
    if (t < 64) S[(size_t)b * D_N + t] = red[t] + red[64 + t];
}

// ---------------------------------------------------------------------------
// Batch-norm stats (training mode, biased var). One block per column.
// Outputs affine form: xn = A[j]*x + B[j]. Deterministic (no atomics).
// ---------------------------------------------------------------------------
__global__ __launch_bounds__(256)
void k_stats(const float* __restrict__ X, const float* __restrict__ g,
             const float* __restrict__ bta, float* __restrict__ A, float* __restrict__ Bo)
{
    __shared__ float rs[256], rq[256];
    int j = blockIdx.x, t = threadIdx.x;
    float s = 0.f, q = 0.f;
    for (int b = t; b < B_N; b += 256) {
        float x = X[(size_t)b * D_N + j];
        s += x; q += x * x;
    }
    rs[t] = s; rq[t] = q;
    __syncthreads();
    for (int st = 128; st > 0; st >>= 1) {
        if (t < st) { rs[t] += rs[t + st]; rq[t] += rq[t + st]; }
        __syncthreads();
    }
    if (t == 0) {
        float m = rs[0] / (float)B_N;
        float v = rq[0] / (float)B_N - m * m;
        float a = rsqrtf(v + BN_EPS) * g[j];
        A[j] = a;
        Bo[j] = bta[j] - m * a;
    }
}

// ---------------------------------------------------------------------------
// Y = selu( BN(S) @ inproj_w + inproj_b ).  4 rows per 256-thread block.
// ---------------------------------------------------------------------------
#define SELU_SCALE 1.0507009873554804934193349852946f
#define SELU_ALPHA 1.6732632423543772848170429916717f

__global__ __launch_bounds__(256)
void k_inproj(const float* __restrict__ S, const float* __restrict__ A1,
              const float* __restrict__ B1, const float* __restrict__ W,
              const float* __restrict__ bias, float* __restrict__ Y)
{
    __shared__ float sW[64 * 64];
    __shared__ float sx[4][64];
    __shared__ float sA[64], sB[64], sb[64];
    int t = threadIdx.x;
    for (int i = t; i < 64 * 64; i += 256) sW[i] = W[i];
    if (t < 64) { sA[t] = A1[t]; sB[t] = B1[t]; sb[t] = bias[t]; }
    int r = t >> 6, j = t & 63;
    int b = blockIdx.x * 4 + r;
    __syncthreads();
    sx[r][j] = sA[j] * S[(size_t)b * D_N + j] + sB[j];
    __syncthreads();
    float acc = sb[j];
    #pragma unroll 8
    for (int k = 0; k < 64; ++k) acc += sx[r][k] * sW[k * 64 + j];
    Y[(size_t)b * D_N + j] = acc > 0.f ? SELU_SCALE * acc
                                       : SELU_SCALE * SELU_ALPHA * (__expf(acc) - 1.f);
}

// ---------------------------------------------------------------------------
// n = BN1(Y) @ outproj + b; self = u2e[nodes];
// beta = sigmoid([self,n,self*n] @ gate1_w + gate1_b); out = mix.
// ---------------------------------------------------------------------------
__global__ __launch_bounds__(256)
void k_final(const float* __restrict__ Y, const float* __restrict__ A2,
             const float* __restrict__ B2, const float* __restrict__ Wout,
             const float* __restrict__ bout, const int* __restrict__ nodes,
             const float* __restrict__ u2e, const float* __restrict__ Wg,
             const float* __restrict__ bg, float* __restrict__ out)
{
    __shared__ float   sWo[64 * 64];
    __shared__ _Float16 sWg[TD * 64];       // gate1_w in f16 to fit LDS
    __shared__ float sA[64], sB[64], sbo[64], sbg[64];
    __shared__ float syn[4][64];
    __shared__ float scomb[4][TD];
    int t = threadIdx.x;
    for (int i = t; i < 64 * 64; i += 256) sWo[i] = Wout[i];
    for (int i = t; i < TD * 64; i += 256) sWg[i] = (_Float16)Wg[i];
    if (t < 64) { sA[t] = A2[t]; sB[t] = B2[t]; sbo[t] = bout[t]; sbg[t] = bg[t]; }
    int r = t >> 6, j = t & 63;
    int b = blockIdx.x * 4 + r;
    __syncthreads();
    syn[r][j] = sA[j] * Y[(size_t)b * D_N + j] + sB[j];
    __syncthreads();
    float n = sbo[j];
    #pragma unroll 8
    for (int k = 0; k < 64; ++k) n += syn[r][k] * sWo[k * 64 + j];
    float self = u2e[(size_t)nodes[b] * D_N + j];
    scomb[r][j]        = self;
    scomb[r][64 + j]   = n;
    scomb[r][128 + j]  = self * n;
    __syncthreads();
    float z = sbg[j];
    #pragma unroll 8
    for (int k = 0; k < TD; ++k) z += scomb[r][k] * (float)sWg[k * 64 + j];
    float beta = 1.f / (1.f + __expf(-z));
    out[(size_t)b * D_N + j] = beta * self + (1.f - beta) * n;
}

// ---------------------------------------------------------------------------
extern "C" void kernel_launch(void* const* d_in, const int* in_sizes, int n_in,
                              void* d_out, int out_size, void* d_ws, size_t ws_size,
                              hipStream_t stream)
{
    (void)in_sizes; (void)n_in; (void)out_size; (void)ws_size;

    const int*   nodes     = (const int*)  d_in[0];
    const int*   hui       = (const int*)  d_in[1];
    const int*   hr        = (const int*)  d_in[2];
    const float* i2e       = (const float*)d_in[3];
    const float* u2e       = (const float*)d_in[4];
    const float* r2e       = (const float*)d_in[5];
    // d_in[6..13]: att1/att2/att3/lin1 — mathematically dead (entmax over size-1 axis == 1)
    const float* gate_w    = (const float*)d_in[14];
    const float* gate_b    = (const float*)d_in[15];
    const float* gate1_w   = (const float*)d_in[16];
    const float* gate1_b   = (const float*)d_in[17];
    const float* bn_g      = (const float*)d_in[18];
    const float* bn_b      = (const float*)d_in[19];
    const float* inproj_w  = (const float*)d_in[20];
    const float* inproj_b  = (const float*)d_in[21];
    const float* bn1_g     = (const float*)d_in[22];
    const float* bn1_b     = (const float*)d_in[23];
    const float* outproj_w = (const float*)d_in[24];
    const float* outproj_b = (const float*)d_in[25];

    float* out = (float*)d_out;
    float* ws  = (float*)d_ws;
    float* S   = ws;                       // [B,64]
    float* Y   = ws + (size_t)B_N * D_N;   // [B,64]
    float* A1  = ws + (size_t)2 * B_N * D_N;
    float* B1  = A1 + 64;
    float* A2  = B1 + 64;
    float* B2  = A2 + 64;

    hipLaunchKernelGGL(k_gate_agg, dim3(B_N),     dim3(128), 0, stream,
                       hui, hr, i2e, r2e, gate_w, gate_b, S);
    hipLaunchKernelGGL(k_stats,    dim3(D_N),     dim3(256), 0, stream,
                       S, bn_g, bn_b, A1, B1);
    hipLaunchKernelGGL(k_inproj,   dim3(B_N / 4), dim3(256), 0, stream,
                       S, A1, B1, inproj_w, inproj_b, Y);
    hipLaunchKernelGGL(k_stats,    dim3(D_N),     dim3(256), 0, stream,
                       Y, bn1_g, bn1_b, A2, B2);
    hipLaunchKernelGGL(k_final,    dim3(B_N / 4), dim3(256), 0, stream,
                       Y, A2, B2, outproj_w, outproj_b, nodes, u2e, gate1_w, gate1_b, out);
}
